// SlidingWindowAttention_73985106641325
// MI455X (gfx1250) — matrix operand order repaired and checked by reading, not compiled, not run on hardware
//
#include <hip/hip_runtime.h>

// ---------------------------------------------------------------------------
// Sliding-window attention for MI455X (gfx1250, wave32, WMMA).
// Pipeline:
//   K0: fp32 -> bf16 conversion of x, qkv_w, out_w            (bandwidth pass)
//   K1: QKV GEMM (bf16 WMMA, f32 acc) + fused RoPE epilogue,
//       scatter to Qh[b,h,t,d], Kh[b,h,t,d], VT[b,h,d,t]
//   K2: banded flash attention, one wave per (b,h,16-query tile):
//       S^T = K*Q^T via v_wmma_f32_16x16x32_bf16 (C-layout of S^T == A-layout
//       of P for the P*V WMMA -> no transpose shuffles needed),
//       online softmax, O += P*V (VT gives contiguous B-operand loads)
//   K3: output projection GEMM (bf16 WMMA, f32 acc) -> fp32 d_out
// ---------------------------------------------------------------------------

typedef __bf16 bf16_t;
typedef __attribute__((ext_vector_type(16))) __bf16 v16bf;
typedef __attribute__((ext_vector_type(8)))  __bf16 v8bf;
typedef __attribute__((ext_vector_type(4)))  __bf16 v4bf;
typedef __attribute__((ext_vector_type(8)))  float  v8f;

constexpr int BATCH = 4;
constexpr int SEQ   = 2048;
constexpr int DM    = 1024;
constexpr int NH    = 16;
constexpr int DH    = 64;
constexpr int WIN   = 256;
constexpr int ROWS  = BATCH * SEQ;   // 8192
constexpr int QKVF  = 3 * DM;        // 3072

// ---- workspace layout (bytes), total ~88 MB of bf16 staging ----
constexpr size_t OFF_XBF  = 0;                                    // [8192,1024]
constexpr size_t OFF_WQKV = OFF_XBF  + (size_t)ROWS * DM * 2;     // [3072,1024]
constexpr size_t OFF_WOUT = OFF_WQKV + (size_t)QKVF * DM * 2;     // [1024,1024]
constexpr size_t OFF_QH   = OFF_WOUT + (size_t)DM * DM * 2;       // [B,H,T,64]
constexpr size_t OFF_KH   = OFF_QH   + (size_t)BATCH * NH * SEQ * DH * 2;
constexpr size_t OFF_VT   = OFF_KH   + (size_t)BATCH * NH * SEQ * DH * 2; // [B,H,64,T]
constexpr size_t OFF_AO   = OFF_VT   + (size_t)BATCH * NH * SEQ * DH * 2; // [8192,1024]

// ---------------------------------------------------------------------------
// Fragment loaders (layouts per CDNA5 ISA 7.12.2, bf16 16x16x32).
// A (16xK=32): lane<16 holds row M=lane, K {0..7,16..23}; lane>=16: K {8..15,24..31}
// B (K=32x16): lane<16 holds col N=lane, K {0..7,16..23}; lane>=16: K {8..15,24..31}
// Both operands come from K-contiguous rows -> pairs of b128 loads.
// ---------------------------------------------------------------------------
__device__ __forceinline__ v16bf load_frag_a(const bf16_t* rowp, int k, int asel) {
  const v8bf lo = *reinterpret_cast<const v8bf*>(rowp + k + asel);
  const v8bf hi = *reinterpret_cast<const v8bf*>(rowp + k + 16 + asel);
  v16bf f;
#pragma unroll
  for (int e = 0; e < 8; ++e) { f[e] = lo[e]; f[e + 8] = hi[e]; }
  return f;
}

__device__ __forceinline__ v16bf load_frag_b(const bf16_t* colp, int k, int bsel) {
  const v8bf lo = *reinterpret_cast<const v8bf*>(colp + k + bsel);
  const v8bf hi = *reinterpret_cast<const v8bf*>(colp + k + 16 + bsel);
  v16bf f;
#pragma unroll
  for (int e = 0; e < 8; ++e) { f[e] = lo[e]; f[e + 8] = hi[e]; }
  return f;
}

__device__ __forceinline__ v8f wmma_bf16(v16bf a, v16bf b, v8f c) {
  // (neg_a, A, neg_b, B, c_mod, C, reuse_a, reuse_b)
  return __builtin_amdgcn_wmma_f32_16x16x32_bf16(false, a, false, b, (short)0, c,
                                                 false, false);
}

// ---------------------------------------------------------------------------
// K0: fp32 -> bf16, vectorized x4
// ---------------------------------------------------------------------------
__global__ void __launch_bounds__(256) swa_cvt_bf16(const float4* __restrict__ in,
                                                    v4bf* __restrict__ out, int n4) {
  int i = blockIdx.x * blockDim.x + threadIdx.x;
  if (i < n4) {
    float4 v = in[i];
    v4bf o;
    o[0] = (bf16_t)v.x; o[1] = (bf16_t)v.y; o[2] = (bf16_t)v.z; o[3] = (bf16_t)v.w;
    out[i] = o;
  }
}

// ---------------------------------------------------------------------------
// K1: QKV GEMM (rows 16 x cols 64 per wave) + fused RoPE + head scatter.
// qkv[m,n] = sum_k Xbf[m,k] * Wqkv[n,k]   (y = x @ W^T)
// ---------------------------------------------------------------------------
__global__ void __launch_bounds__(256) swa_qkv_rope(
    const bf16_t* __restrict__ Xbf, const bf16_t* __restrict__ Wqkv,
    const float* __restrict__ cosp, const float* __restrict__ sinp,
    bf16_t* __restrict__ Qh, bf16_t* __restrict__ Kh, bf16_t* __restrict__ VT)
{
  const int lane = threadIdx.x & 31;
  const int wid  = (blockIdx.x * blockDim.x + threadIdx.x) >> 5;
  const int mt = wid & 511;      // 512 row tiles of 16
  const int nt = wid >> 9;       // 48 col tiles of 64
  if (nt >= QKVF / 64) return;

  const int cl     = lane & 15;
  const int rowoff = (lane >> 4) << 3;
  const int asel   = (lane < 16) ? 0 : 8;
  const int bsel   = (lane < 16) ? 0 : 8;
  const int mbase  = mt << 4;
  const int nbase  = nt << 6;

  const bf16_t* arow = Xbf + (size_t)(mbase + cl) * DM;
  const bf16_t* brow[4];
#pragma unroll
  for (int c = 0; c < 4; ++c) brow[c] = Wqkv + (size_t)(nbase + c * 16 + cl) * DM;

  v8f acc[4] = {};
  for (int k = 0; k < DM; k += 32) {
    __builtin_prefetch(arow + k + 64, 0, 1);
    const v16bf a = load_frag_a(arow, k, asel);
#pragma unroll
    for (int c = 0; c < 4; ++c)
      acc[c] = wmma_bf16(a, load_frag_b(brow[c], k, bsel), acc[c]);
  }

  // which of q/k/v, head, (b, t)
  const int which = nt >> 4;          // 16 head-tiles per 1024-wide block
  const int h     = nt & 15;
  const int b     = mt >> 7;          // 128 row tiles per batch
  const int tb    = (mt & 127) << 4;

  if (which < 2) {
    // RoPE: y[d] = x[d]*cos + rot[d]*sin,  rot[d] = (d<32 ? -x[d+32] : x[d-32])
    // chunk c and c±2 live in the same lane/register -> register-local rotate.
    bf16_t* dst = (which == 0 ? Qh : Kh) + (size_t)(b * NH + h) * SEQ * DH;
#pragma unroll
    for (int c = 0; c < 4; ++c) {
      const int d = c * 16 + cl;
#pragma unroll
      for (int r = 0; r < 8; ++r) {
        const int   t  = tb + r + rowoff;
        const float x  = acc[c][r];
        const float xr = (c < 2) ? -acc[c + 2][r] : acc[c - 2][r];
        const float y  = x * cosp[t * DH + d] + xr * sinp[t * DH + d];
        dst[(size_t)t * DH + d] = (bf16_t)y;
      }
    }
  } else {
    // V stored transposed: VT[b,h,d,t] so the PV B-operand is K-contiguous.
    bf16_t* dst = VT + (size_t)(b * NH + h) * DH * SEQ;
#pragma unroll
    for (int c = 0; c < 4; ++c) {
      const int d = c * 16 + cl;
#pragma unroll
      for (int r = 0; r < 8; ++r) {
        const int t = tb + r + rowoff;
        dst[(size_t)d * SEQ + t] = (bf16_t)acc[c][r];
      }
    }
  }
}

// ---------------------------------------------------------------------------
// K2: banded flash attention. One wave handles one (b, h, 16-query tile).
// S^T = K @ Q^T so the exp'd scores land directly in the A-operand layout
// of the P@V WMMA (no LDS / shuffle transpose).
// ---------------------------------------------------------------------------
__global__ void __launch_bounds__(256) swa_attention(
    const bf16_t* __restrict__ Qh, const bf16_t* __restrict__ Kh,
    const bf16_t* __restrict__ VT, bf16_t* __restrict__ AttnOut)
{
  const int lane = threadIdx.x & 31;
  const int wid  = (blockIdx.x * blockDim.x + threadIdx.x) >> 5;
  const int qt = wid & 127;
  const int h  = (wid >> 7) & 15;
  const int b  = wid >> 11;
  if (b >= BATCH) return;

  const int cl     = lane & 15;
  const int rowoff = (lane >> 4) << 3;
  const int asel   = (lane < 16) ? 0 : 8;
  const int bsel   = (lane < 16) ? 0 : 8;
  const int i0     = qt << 4;
  const int qidx   = i0 + cl;          // this lane's query index

  const size_t bh = (size_t)(b * NH + h);
  const bf16_t* Qb = Qh + bh * SEQ * DH;
  const bf16_t* Kb = Kh + bh * SEQ * DH;
  const bf16_t* Vb = VT + bh * DH * SEQ;

  // Q is the B operand of S^T (columns = queries, K-dim = d)
  const bf16_t* qrow = Qb + (size_t)(i0 + cl) * DH;
  const v16bf qf0 = load_frag_b(qrow, 0, bsel);
  const v16bf qf1 = load_frag_b(qrow, 32, bsel);

  v8f   o[4] = {};
  float m = -1e30f, lsum = 0.0f;
  const float scale = 0.125f;          // 1/sqrt(64)
  const int ks = (i0 > (WIN - 1)) ? ((i0 - (WIN - 1)) & ~31) : 0;

  for (int kb = ks; kb <= i0 + 15; kb += 32) {   // kb <= T-32 always holds
    const bf16_t* kr0 = Kb + (size_t)(kb + cl) * DH;
    const bf16_t* kr1 = Kb + (size_t)(kb + 16 + cl) * DH;
    v8f st0 = {}, st1 = {};
    st0 = wmma_bf16(load_frag_a(kr0, 0,  asel), qf0, st0);
    st0 = wmma_bf16(load_frag_a(kr0, 32, asel), qf1, st0);
    st1 = wmma_bf16(load_frag_a(kr1, 0,  asel), qf0, st1);
    st1 = wmma_bf16(load_frag_a(kr1, 32, asel), qf1, st1);

    float s0[8], s1[8]; bool ok0[8], ok1[8];
    float bm = -1e30f;
#pragma unroll
    for (int r = 0; r < 8; ++r) {
      const int k0 = kb + r + rowoff;        // key index of st0[r] in this lane
      const int k1 = k0 + 16;
      s0[r]  = st0[r] * scale;
      s1[r]  = st1[r] * scale;
      ok0[r] = (k0 <= qidx) && (k0 >= qidx - (WIN - 1));
      ok1[r] = (k1 <= qidx) && (k1 >= qidx - (WIN - 1));
      if (ok0[r]) bm = fmaxf(bm, s0[r]);
      if (ok1[r]) bm = fmaxf(bm, s1[r]);
    }
    bm = fmaxf(bm, __shfl_xor(bm, 16, 32));  // lanes q and q+16 share a query
    const float mnew  = fmaxf(m, bm);
    const float alpha = __expf(m - mnew);
    m = mnew;

    // exp'd probabilities pack straight into the P@V A-fragment.
    float psum = 0.0f;
    v16bf ap;
#pragma unroll
    for (int r = 0; r < 8; ++r) {
      const float p0 = ok0[r] ? __expf(s0[r] - mnew) : 0.0f;
      const float p1 = ok1[r] ? __expf(s1[r] - mnew) : 0.0f;
      psum += p0 + p1;
      ap[r]     = (bf16_t)p0;
      ap[r + 8] = (bf16_t)p1;
    }
    psum += __shfl_xor(psum, 16, 32);
    lsum  = lsum * alpha + psum;

    // rescale O: row of O-accum r corresponds to query (r + rowoff)
    float arw[8];
#pragma unroll
    for (int r = 0; r < 8; ++r) arw[r] = __shfl(alpha, r + rowoff, 32);
#pragma unroll
    for (int c = 0; c < 4; ++c)
#pragma unroll
      for (int r = 0; r < 8; ++r) o[c][r] *= arw[r];

#pragma unroll
    for (int c = 0; c < 4; ++c) {
      const bf16_t* vrow = Vb + (size_t)(c * 16 + cl) * SEQ;
      o[c] = wmma_bf16(ap, load_frag_b(vrow, kb, bsel), o[c]);
    }
  }

  const float linv = 1.0f / lsum;
  float lr[8];
#pragma unroll
  for (int r = 0; r < 8; ++r) lr[r] = __shfl(linv, r + rowoff, 32);

  bf16_t* dst = AttnOut + (size_t)b * SEQ * DM + (size_t)h * DH;
#pragma unroll
  for (int c = 0; c < 4; ++c)
#pragma unroll
    for (int r = 0; r < 8; ++r)
      dst[(size_t)(i0 + r + rowoff) * DM + c * 16 + cl] = (bf16_t)(o[c][r] * lr[r]);
}

// ---------------------------------------------------------------------------
// K3: output projection  out[m,n] = sum_k AttnOut[m,k] * Wout[n,k]  (fp32 out)
// ---------------------------------------------------------------------------
__global__ void __launch_bounds__(256) swa_out_proj(
    const bf16_t* __restrict__ A, const bf16_t* __restrict__ W,
    float* __restrict__ out)
{
  const int lane = threadIdx.x & 31;
  const int wid  = (blockIdx.x * blockDim.x + threadIdx.x) >> 5;
  const int mt = wid & 511;
  const int nt = wid >> 9;
  if (nt >= DM / 64) return;

  const int cl     = lane & 15;
  const int rowoff = (lane >> 4) << 3;
  const int asel   = (lane < 16) ? 0 : 8;
  const int bsel   = (lane < 16) ? 0 : 8;
  const int mbase  = mt << 4;
  const int nbase  = nt << 6;

  const bf16_t* arow = A + (size_t)(mbase + cl) * DM;
  const bf16_t* brow[4];
#pragma unroll
  for (int c = 0; c < 4; ++c) brow[c] = W + (size_t)(nbase + c * 16 + cl) * DM;

  v8f acc[4] = {};
  for (int k = 0; k < DM; k += 32) {
    __builtin_prefetch(arow + k + 64, 0, 1);
    const v16bf a = load_frag_a(arow, k, asel);
#pragma unroll
    for (int c = 0; c < 4; ++c)
      acc[c] = wmma_bf16(a, load_frag_b(brow[c], k, bsel), acc[c]);
  }

#pragma unroll
  for (int c = 0; c < 4; ++c)
#pragma unroll
    for (int r = 0; r < 8; ++r)
      out[(size_t)(mbase + r + rowoff) * DM + nbase + c * 16 + cl] = acc[c][r];
}

// ---------------------------------------------------------------------------
extern "C" void kernel_launch(void* const* d_in, const int* in_sizes, int n_in,
                              void* d_out, int out_size, void* d_ws, size_t ws_size,
                              hipStream_t stream) {
  const float* x     = (const float*)d_in[0];
  const float* cosp  = (const float*)d_in[1];
  const float* sinp  = (const float*)d_in[2];
  const float* qkv_w = (const float*)d_in[3];
  const float* out_w = (const float*)d_in[4];
  float* out = (float*)d_out;

  char* ws = (char*)d_ws;
  bf16_t* Xbf  = (bf16_t*)(ws + OFF_XBF);
  bf16_t* Wqkv = (bf16_t*)(ws + OFF_WQKV);
  bf16_t* Wout = (bf16_t*)(ws + OFF_WOUT);
  bf16_t* Qh   = (bf16_t*)(ws + OFF_QH);
  bf16_t* Kh   = (bf16_t*)(ws + OFF_KH);
  bf16_t* VT   = (bf16_t*)(ws + OFF_VT);
  bf16_t* AO   = (bf16_t*)(ws + OFF_AO);

  const int nX  = ROWS * DM;      // 8.4M
  const int nWq = QKVF * DM;      // 3.1M
  const int nWo = DM * DM;        // 1.0M

  swa_cvt_bf16<<<(nX / 4 + 255) / 256, 256, 0, stream>>>(
      (const float4*)x, (v4bf*)Xbf, nX / 4);
  swa_cvt_bf16<<<(nWq / 4 + 255) / 256, 256, 0, stream>>>(
      (const float4*)qkv_w, (v4bf*)Wqkv, nWq / 4);
  swa_cvt_bf16<<<(nWo / 4 + 255) / 256, 256, 0, stream>>>(
      (const float4*)out_w, (v4bf*)Wout, nWo / 4);

  // 512 row-tiles * 48 col-tiles = 24576 waves, 8 waves/block
  swa_qkv_rope<<<3072, 256, 0, stream>>>(Xbf, Wqkv, cosp, sinp, Qh, Kh, VT);

  // 4*16*128 = 8192 waves
  swa_attention<<<1024, 256, 0, stream>>>(Qh, Kh, VT, AO);

  // 512 * 16 = 8192 waves
  swa_out_proj<<<1024, 256, 0, stream>>>(AO, Wout, out);
}
